// Decoder_67551245631959
// MI455X (gfx1250) — matrix operand order repaired and checked by reading, not compiled, not run on hardware
//
#include <hip/hip_runtime.h>
#include <hip/hip_bf16.h>

typedef __attribute__((ext_vector_type(16))) _Float16 v16h;
typedef __attribute__((ext_vector_type(8)))  _Float16 v8h;
typedef __attribute__((ext_vector_type(8)))  float    v8f;

#define TM 256
#define TN 64
#define TK 32
#define EPS 1e-5f
#define NCLS 8

// -------------------------------------------------------------------------
// Weight transform: f32 OIHW [Cout][Cin][3][3] -> f16 [Cout][9*Cin],
// K ordered kk = (ky*3+kx)*Cin + ci so 32-wide K chunks are contiguous ci.
// -------------------------------------------------------------------------
__global__ void wprep_kernel(const float* __restrict__ w, _Float16* __restrict__ wT,
                             int Cout, int Cin) {
    int idx = blockIdx.x * blockDim.x + threadIdx.x;
    int total = Cout * Cin * 9;
    if (idx >= total) return;
    int co = idx / (Cin * 9);
    int r  = idx - co * Cin * 9;
    int t9 = r / Cin;              // ky*3+kx
    int ci = r - t9 * Cin;
    wT[idx] = (_Float16)w[(size_t)(co * Cin + ci) * 9 + t9];
}

// f32 NCHW -> f16 NHWC
__global__ void nchw2nhwc_kernel(const float* __restrict__ in, _Float16* __restrict__ out,
                                 int B, int C, int H, int W) {
    size_t idx = (size_t)blockIdx.x * blockDim.x + threadIdx.x;
    size_t total = (size_t)B * H * W * C;
    if (idx >= total) return;
    int ch = (int)(idx % C);
    size_t m = idx / C;
    int b = (int)(m / ((size_t)H * W));
    int p = (int)(m - (size_t)b * H * W);
    int y = p / W, x = p - y * W;
    out[idx] = (_Float16)in[(((size_t)b * C + ch) * H + y) * W + x];
}

// -------------------------------------------------------------------------
// Implicit-GEMM 3x3 reflect-pad conv, NHWC f16 activations, f32 accum WMMA.
// Block = 256 threads = 8 waves; tile 256M x 64N, K-step 32.
// K loop is tap-outer (9 taps, reflect math hoisted) / channel-inner
// (pure pointer increments). Each wave: 32M x 64N patch via 2x4 fragments
// of v_wmma_f32_16x16x32_f16 (8 WMMA per K-step).
// -------------------------------------------------------------------------
__global__ __launch_bounds__(256)
void conv3x3_wmma_kernel(const _Float16* __restrict__ in,   // NHWC f16
                         const _Float16* __restrict__ wT,   // [Cout][9*Cin]
                         const float*    __restrict__ bias, // [Cout] f32
                         _Float16* __restrict__ out,        // NHWC f16
                         int H, int W, int Cin, int Cout, int relu) {
    __shared__ _Float16 As[TM * TK];   // A tile [m][k]
    __shared__ _Float16 Bs[TN * TK];   // B tile transposed [n][k]

    const int tid   = threadIdx.x;
    const int lane  = tid & 31;
    const int waveM = tid >> 5;        // 0..7, wave tile = 32M x 64N
    const int g     = lane >> 4;       // lane group
    const int lr    = lane & 15;

    const int m0   = blockIdx.x * TM;
    const int n0   = blockIdx.y * TN;
    const int HW   = H * W;
    const int Ktot = 9 * Cin;

    // A loader: thread stages its entire 32-half row (two 32B copies)
    const int am   = m0 + tid;
    const int ab   = am / HW;
    const int arem = am - ab * HW;
    const int ay   = arem / W;
    const int ax   = arem - ay * W;
    const _Float16* aPix = in + (size_t)ab * HW * Cin;   // batch base

    // B loader: thread handles n-row (tid>>2), 8-half chunk ((tid&3)*8)
    const int bRow = tid >> 2;
    const int bCh  = (tid & 3) * 8;
    const _Float16* wRow = wT + (size_t)(n0 + bRow) * Ktot + bCh;

    v8f c[2][4];
    #pragma unroll
    for (int f = 0; f < 2; ++f)
        #pragma unroll
        for (int j = 0; j < 4; ++j)
            c[f][j] = (v8f){0.f};

    for (int t9 = 0; t9 < 9; ++t9) {
        // ---- hoisted per-tap reflect-pad addressing ----
        const int ky = t9 / 3;
        const int kx = t9 - 3 * ky;
        int ry = ay + ky - 1; ry = (ry < 0) ? 1 : (ry >= H ? H - 2 : ry);
        int rx = ax + kx - 1; rx = (rx < 0) ? 1 : (rx >= W ? W - 2 : rx);
        const _Float16* srcTap = aPix + ((size_t)ry * W + rx) * Cin;
        const _Float16* wTap   = wRow + (size_t)t9 * Cin;

        for (int ci = 0; ci < Cin; ci += TK) {
            // ---- stage A row: 64B contiguous per thread ----
            *(v16h*)(As + tid * TK)      = *(const v16h*)(srcTap + ci);
            *(v16h*)(As + tid * TK + 16) = *(const v16h*)(srcTap + ci + 16);
            // ---- stage B tile (transposed: Bs[n][k], k contiguous) ----
            *(v8h*)(Bs + bRow * TK + bCh) = *(const v8h*)(wTap + ci);
            if (ci + TK < Cin) {
                __builtin_prefetch(srcTap + ci + TK, 0, 0);  // global_prefetch_b8
                __builtin_prefetch(wTap + ci + TK, 0, 0);
            }

            __syncthreads();

            // ---- fragment loads: contiguous-K order (k = 16*g + s), same for A and B ----
            // A 16x32 f16: lane holds 16 contiguous K at g*16
            v16h a[2], b[4];
            #pragma unroll
            for (int f = 0; f < 2; ++f) {
                const _Float16* ap = As + (waveM * 32 + f * 16 + lr) * TK;
                v8h lo = *(const v8h*)(ap + g * 16);
                v8h hi = *(const v8h*)(ap + g * 16 + 8);
                a[f] = __builtin_shufflevector(lo, hi,
                         0,1,2,3,4,5,6,7,8,9,10,11,12,13,14,15);
            }
            // B 32x16 f16: lane = N column, 16 contiguous K at g*16
            #pragma unroll
            for (int j = 0; j < 4; ++j)
                b[j] = *(const v16h*)(Bs + (j * 16 + lr) * TK + g * 16);

            #pragma unroll
            for (int f = 0; f < 2; ++f)
                #pragma unroll
                for (int j = 0; j < 4; ++j)
                    c[f][j] = __builtin_amdgcn_wmma_f32_16x16x32_f16(
                        false, a[f], false, b[j], (short)0, c[f][j], false, false);

            __syncthreads();
        }
    }

    // ---- epilogue: C layout M = r + g*8, N = lane&15; bias + relu ----
    const int mBase = m0 + waveM * 32;
    #pragma unroll
    for (int f = 0; f < 2; ++f) {
        #pragma unroll
        for (int j = 0; j < 4; ++j) {
            int ng = n0 + j * 16 + lr;
            float bv = bias[ng];
            int mg = mBase + f * 16 + g * 8;
            #pragma unroll
            for (int r = 0; r < 8; ++r) {
                float v = c[f][j][r] + bv;
                if (relu) v = fmaxf(v, 0.f);
                out[(size_t)(mg + r) * Cout + ng] = (_Float16)v;
            }
        }
    }
}

// -------------------------------------------------------------------------
// Seg-norm. out = x * rs[class(p)] - sum_c mu_c*rs_c  (per b,ch).
// Stats: one block per (b,ch), 8 class sums + sumsqs at once.
// -------------------------------------------------------------------------
__global__ __launch_bounds__(256)
void segstats_kernel(const _Float16* __restrict__ act, const int* __restrict__ seg,
                     float* __restrict__ rsArr, float* __restrict__ Sarr,
                     int C, int H, int W) {
    int bc = blockIdx.x;
    int b  = bc / C;
    int ch = bc - b * C;
    int HW = H * W;
    float s[NCLS] = {0}, q[NCLS] = {0};
    for (int p = threadIdx.x; p < HW; p += 256) {
        int y = p / W, x = p - y * W;
        int iy = (y * 64) / H;           // nearest resize of 64x64 seg
        int ix = (x * 64) / W;
        int k = seg[((size_t)b * 64 + iy) * 64 + ix];
        float v = (float)act[((size_t)b * HW + p) * C + ch];
        s[k] += v; q[k] += v * v;
    }
    __shared__ float red[256];
    __shared__ float totals[2 * NCLS];
    for (int qi = 0; qi < 2 * NCLS; ++qi) {
        red[threadIdx.x] = (qi < NCLS) ? s[qi] : q[qi - NCLS];
        __syncthreads();
        for (int st = 128; st > 0; st >>= 1) {
            if (threadIdx.x < st) red[threadIdx.x] += red[threadIdx.x + st];
            __syncthreads();
        }
        if (threadIdx.x == 0) totals[qi] = red[0];
        __syncthreads();
    }
    if (threadIdx.x == 0) {
        float S = 0.f, inv = 1.0f / (float)HW;
        for (int k = 0; k < NCLS; ++k) {
            float mu  = totals[k] * inv;
            float var = totals[NCLS + k] * inv - mu * mu;
            float rs  = rsqrtf(var + EPS);
            rsArr[bc * NCLS + k] = rs;
            S += mu * rs;
        }
        Sarr[bc] = S;
    }
}

__global__ void segapply_kernel(_Float16* __restrict__ act, const int* __restrict__ seg,
                                const float* __restrict__ rsArr, const float* __restrict__ Sarr,
                                int C, int H, int W) {
    size_t idx = (size_t)blockIdx.x * blockDim.x + threadIdx.x;
    size_t total = (size_t)4 * H * W * C;
    if (idx >= total) return;
    int ch = (int)(idx % C);
    size_t m = idx / C;
    int HW = H * W;
    int b = (int)(m / HW);
    int p = (int)(m - (size_t)b * HW);
    int y = p / W, x = p - y * W;
    int iy = (y * 64) / H, ix = (x * 64) / W;
    int k = seg[((size_t)b * 64 + iy) * 64 + ix];
    int bc = b * C + ch;
    float v = (float)act[idx];
    act[idx] = (_Float16)(v * rsArr[bc * NCLS + k] - Sarr[bc]);
}

// -------------------------------------------------------------------------
// 2x bilinear upsample, half-pixel centers (jax.image.resize 'bilinear').
// NHWC f16, 8-channel vectors per thread.
// -------------------------------------------------------------------------
__global__ void up2_kernel(const _Float16* __restrict__ in, _Float16* __restrict__ out,
                           int C, int H, int W) {
    int OH = 2 * H, OW = 2 * W, C8 = C >> 3;
    size_t idx = (size_t)blockIdx.x * blockDim.x + threadIdx.x;
    size_t total = (size_t)4 * OH * OW * C8;
    if (idx >= total) return;
    int c8 = (int)(idx % C8);
    size_t t = idx / C8;
    int ox = (int)(t % OW); t /= OW;
    int oy = (int)(t % OH);
    int b  = (int)(t / OH);

    float sy = oy * 0.5f - 0.25f; if (sy < 0.f) sy = 0.f;
    float sx = ox * 0.5f - 0.25f; if (sx < 0.f) sx = 0.f;
    int y0 = (int)sy; float fy = sy - y0; int y1 = y0 + 1; if (y1 > H - 1) y1 = H - 1;
    int x0 = (int)sx; float fx = sx - x0; int x1 = x0 + 1; if (x1 > W - 1) x1 = W - 1;

    const _Float16* base = in + (size_t)b * H * W * C + (size_t)c8 * 8;
    v8h p00 = *(const v8h*)(base + ((size_t)y0 * W + x0) * C);
    v8h p01 = *(const v8h*)(base + ((size_t)y0 * W + x1) * C);
    v8h p10 = *(const v8h*)(base + ((size_t)y1 * W + x0) * C);
    v8h p11 = *(const v8h*)(base + ((size_t)y1 * W + x1) * C);
    v8h o;
    #pragma unroll
    for (int i = 0; i < 8; ++i) {
        float top = (float)p00[i] * (1.f - fx) + (float)p01[i] * fx;
        float bot = (float)p10[i] * (1.f - fx) + (float)p11[i] * fx;
        o[i] = (_Float16)(top * (1.f - fy) + bot * fy);
    }
    *(v8h*)(out + (((size_t)b * OH + oy) * OW + ox) * C + (size_t)c8 * 8) = o;
}

// -------------------------------------------------------------------------
// Final 64->3 conv (3.6 GFLOP, VALU is fine): NHWC f16 in, f32 NCHW out.
// -------------------------------------------------------------------------
__global__ void convf_kernel(const _Float16* __restrict__ in, const float* __restrict__ w,
                             const float* __restrict__ bias, float* __restrict__ out,
                             int H, int W, int Cin) {
    size_t idx = (size_t)blockIdx.x * blockDim.x + threadIdx.x;
    size_t total = (size_t)4 * 3 * H * W;
    if (idx >= total) return;
    int x = (int)(idx % W);
    size_t t = idx / W;
    int y  = (int)(t % H); t /= H;
    int co = (int)(t % 3);
    int b  = (int)(t / 3);
    float acc = bias[co];
    for (int ky = 0; ky < 3; ++ky) {
        int ry = y + ky - 1; ry = (ry < 0) ? 1 : (ry >= H ? H - 2 : ry);
        for (int kx = 0; kx < 3; ++kx) {
            int rx = x + kx - 1; rx = (rx < 0) ? 1 : (rx >= W ? W - 2 : rx);
            const _Float16* src = in + (((size_t)b * H + ry) * W + rx) * Cin;
            const float* wp = w + (size_t)co * Cin * 9 + ky * 3 + kx;
            for (int ci = 0; ci < Cin; ++ci)
                acc += (float)src[ci] * wp[(size_t)ci * 9];
        }
    }
    out[idx] = acc;
}

// -------------------------------------------------------------------------
extern "C" void kernel_launch(void* const* d_in, const int* in_sizes, int n_in,
                              void* d_out, int out_size, void* d_ws, size_t ws_size,
                              hipStream_t stream) {
    const float* x   = (const float*)d_in[0];
    const int*   seg = (const int*)d_in[1];
    const float* W_[9]; const float* B_[9];
    for (int i = 0; i < 9; ++i) { W_[i] = (const float*)d_in[2 + 2 * i]; B_[i] = (const float*)d_in[3 + 2 * i]; }
    float* outp = (float*)d_out;

    // layer geometry: {Cin, Cout}
    const int CI[8] = {512, 256, 256, 256, 256, 128, 128, 64};
    const int CO[8] = {256, 256, 256, 256, 128, 128,  64, 64};

    // workspace layout
    char* ws = (char*)d_ws;
    size_t off = 0;
    auto alloc = [&](size_t bytes) -> void* {
        void* p = ws + off; off += (bytes + 255) & ~(size_t)255; return p;
    };
    const size_t BIG = (size_t)4 * 512 * 512 * 64 * sizeof(_Float16);  // 134 MB
    _Float16* bufA = (_Float16*)alloc(BIG);
    _Float16* bufB = (_Float16*)alloc(BIG);
    _Float16* wT[8];
    for (int i = 0; i < 8; ++i) wT[i] = (_Float16*)alloc((size_t)CO[i] * CI[i] * 9 * sizeof(_Float16));
    float* rsArr = (float*)alloc((size_t)4 * 256 * NCLS * sizeof(float));
    float* Sarr  = (float*)alloc((size_t)4 * 256 * sizeof(float));

    // 1) weight transforms for the 8 WMMA layers
    for (int i = 0; i < 8; ++i) {
        int total = CO[i] * CI[i] * 9;
        wprep_kernel<<<(total + 255) / 256, 256, 0, stream>>>(W_[i], wT[i], CO[i], CI[i]);
    }
    // 2) input f32 NCHW -> f16 NHWC
    {
        size_t total = (size_t)4 * 512 * 64 * 64;
        nchw2nhwc_kernel<<<(unsigned)((total + 255) / 256), 256, 0, stream>>>(x, bufA, 4, 512, 64, 64);
    }

    auto conv = [&](const _Float16* src, _Float16* dst, int li, int H, int Wd, int relu) {
        int M = 4 * H * Wd;
        dim3 grid(M / TM, CO[li] / TN);
        conv3x3_wmma_kernel<<<grid, 256, 0, stream>>>(src, wT[li], B_[li], dst, H, Wd, CI[li], CO[li], relu);
    };
    auto segnorm = [&](_Float16* a, int C, int H, int Wd) {
        segstats_kernel<<<4 * C, 256, 0, stream>>>(a, seg, rsArr, Sarr, C, H, Wd);
        size_t total = (size_t)4 * H * Wd * C;
        segapply_kernel<<<(unsigned)((total + 255) / 256), 256, 0, stream>>>(a, seg, rsArr, Sarr, C, H, Wd);
    };
    auto up2 = [&](const _Float16* src, _Float16* dst, int C, int H, int Wd) {
        size_t total = (size_t)4 * (2 * H) * (2 * Wd) * (C >> 3);
        up2_kernel<<<(unsigned)((total + 255) / 256), 256, 0, stream>>>(src, dst, C, H, Wd);
    };

    conv(bufA, bufB, 0, 64, 64, 1);          // conv1 512->256 @64
    segnorm(bufB, 256, 64, 64);
    up2(bufB, bufA, 256, 64, 64);            // -> 128x128
    conv(bufA, bufB, 1, 128, 128, 1);        // conv2
    conv(bufB, bufA, 2, 128, 128, 1);        // conv3
    conv(bufA, bufB, 3, 128, 128, 1);        // conv4
    conv(bufB, bufA, 4, 128, 128, 1);        // conv5 256->128
    segnorm(bufA, 128, 128, 128);
    up2(bufA, bufB, 128, 128, 128);          // -> 256x256
    conv(bufB, bufA, 5, 256, 256, 1);        // conv6
    conv(bufA, bufB, 6, 256, 256, 1);        // conv7 128->64
    segnorm(bufB, 64, 256, 256);
    up2(bufB, bufA, 64, 256, 256);           // -> 512x512
    conv(bufA, bufB, 7, 512, 512, 1);        // conv8 64->64
    {
        size_t total = (size_t)4 * 3 * 512 * 512;
        convf_kernel<<<(unsigned)((total + 255) / 256), 256, 0, stream>>>(bufB, W_[8], B_[8], outp, 512, 512, 64);
    }
}
